// Single_Layer_ONN_15942918603115
// MI455X (gfx1250) — matrix operand order repaired and checked by reading, not compiled, not run on hardware
//
#include <hip/hip_runtime.h>
#include <hip/hip_fp16.h>

typedef __attribute__((ext_vector_type(16))) _Float16 v16h;
typedef __attribute__((ext_vector_type(8)))  float    v8f;

// Prologue: convert lin_w [512 x 64] fp32 -> f16 into workspace (64 KB, L2-resident)
__global__ void convert_linw_kernel(const float* __restrict__ lw, __half* __restrict__ out) {
    int i = blockIdx.x * 256 + threadIdx.x;
    if (i < 512 * 64) out[i] = __float2half(lw[i]);
}

// One block per (b, h): Self-ONN conv3x3 (+tanh) for all 64 channels x 64 width,
// then WMMA GEMM [64c x 64w] @ [64w x 512e] with fused bias + L2-normalize over c.
__global__ __launch_bounds__(256)
void onn_fused_kernel(const float* __restrict__ x,        // [16,3,512,64]
                      const float* __restrict__ conv_w,   // [64,9,3,3]
                      const float* __restrict__ conv_b,   // [64]
                      const __half* __restrict__ linw_h,  // [512,64] f16
                      const float* __restrict__ lin_b,    // [512]
                      float* __restrict__ out)            // [16,64,512,512]
{
    __shared__ float s_w[64 * 81];                 // conv weights  (20.25 KB)
    __shared__ float s_xq[3][3][3][66];            // powers p, cin, row, padded w (7 KB)
    __shared__ __align__(16) __half s_a[64 * 64];  // tanh(conv) A-tile [c][w]     (8 KB)

    const int tid = threadIdx.x;
    const int bh  = blockIdx.x;
    const int b   = bh >> 9;     // grid = 16 * 512
    const int h   = bh & 511;

    // ---- stage conv weights into LDS ----
    for (int i = tid; i < 64 * 81; i += 256) s_w[i] = conv_w[i];

    // ---- stage x rows h-1..h+1 (zero-padded) and powers x, x^2, x^3 ----
    for (int i = tid; i < 3 * 3 * 66; i += 256) {
        int wp = i % 66;
        int r  = (i / 66) % 3;
        int ci = i / 198;
        int wg = wp - 1;
        int hg = h - 1 + r;
        float v = 0.0f;
        if (wg >= 0 && wg < 64 && hg >= 0 && hg < 512)
            v = x[((b * 3 + ci) * 512 + hg) * 64 + wg];
        s_xq[0][ci][r][wp] = v;
        s_xq[1][ci][r][wp] = v * v;
        s_xq[2][ci][r][wp] = v * v * v;
    }
    __syncthreads();

    // ---- phase 1: conv + tanh -> f16 A-tile in LDS ----
    {
        const int wcol = tid & 63;
        const int cg   = tid >> 6;          // wave-uniform -> LDS weight broadcast
        float xv[81];
        #pragma unroll
        for (int p = 0; p < 3; ++p)
            #pragma unroll
            for (int ci = 0; ci < 3; ++ci)
                #pragma unroll
                for (int ky = 0; ky < 3; ++ky)
                    #pragma unroll
                    for (int kx = 0; kx < 3; ++kx)
                        xv[((p * 3 + ci) * 3 + ky) * 3 + kx] = s_xq[p][ci][ky][wcol + kx];

        #pragma unroll
        for (int k = 0; k < 16; ++k) {
            int c = cg + 4 * k;
            float acc = conv_b[c];
            const float* wr = &s_w[c * 81];
            #pragma unroll
            for (int t = 0; t < 81; ++t) acc = fmaf(xv[t], wr[t], acc);
            s_a[c * 64 + wcol] = __float2half(tanhf(acc));
        }
    }
    __syncthreads();

    // ---- phase 2: WMMA GEMM + fused bias / L2-normalize(c) / store ----
    const int lane  = tid & 31;
    const int wv    = tid >> 5;       // 8 waves, each owns 64 e-values
    const int lrow  = lane & 15;
    const int g     = lane >> 4;      // half-wave selector
    const int eBase = wv * 64;

    union VH { v16h v; uint4 q[2]; };

    // A fragments (16-bit A 16x32 layout: K = 16*(i/8) + 8*g + i%8), 4 M-tiles x 2 K-steps
    VH afrag[4][2];
    #pragma unroll
    for (int mt = 0; mt < 4; ++mt)
        #pragma unroll
        for (int kk = 0; kk < 2; ++kk) {
            const __half* p = &s_a[(mt * 16 + lrow) * 64 + kk * 32 + g * 8];
            afrag[mt][kk].q[0] = *(const uint4*)p;          // K base .. base+7
            afrag[mt][kk].q[1] = *(const uint4*)(p + 16);   // K base+16 .. base+23
        }

    const long base0 = (long)b * 16777216 + (long)h * 512; // b*64*512*512 + h*512
    const v8f  vzero = {0.f, 0.f, 0.f, 0.f, 0.f, 0.f, 0.f, 0.f};

    #pragma unroll
    for (int nt = 0; nt < 4; ++nt) {
        const int e = eBase + nt * 16 + lrow;

        // B fragments (16-bit B 32x16 layout: K = 16*(i/8) + 8*g + i%8, matching A)
        VH bfrag[2];
        #pragma unroll
        for (int kk = 0; kk < 2; ++kk) {
            const __half* p = linw_h + e * 64 + kk * 32 + g * 8;
            bfrag[kk].q[0] = *(const uint4*)p;
            bfrag[kk].q[1] = *(const uint4*)(p + 16);
        }

        v8f acc[4];
        #pragma unroll
        for (int mt = 0; mt < 4; ++mt) acc[mt] = vzero;

        #pragma unroll
        for (int kk = 0; kk < 2; ++kk) {
            #pragma unroll
            for (int mt = 0; mt < 4; ++mt)
                acc[mt] = __builtin_amdgcn_wmma_f32_16x16x32_f16(
                    false, afrag[mt][kk].v, false, bfrag[kk].v,
                    (short)0, acc[mt], false, false);
        }

        // bias, then sum of squares over c (this lane covers 32 of 64 c's; partner = lane^16)
        const float lb = lin_b[e];
        float part = 0.0f;
        #pragma unroll
        for (int mt = 0; mt < 4; ++mt)
            #pragma unroll
            for (int j = 0; j < 8; ++j) {
                float v = acc[mt][j] + lb;
                acc[mt][j] = v;
                part = fmaf(v, v, part);
            }
        float tot   = part + __shfl_xor(part, 16, 32);
        float scale = 1.0f / fmaxf(sqrtf(tot), 1e-12f);

        #pragma unroll
        for (int mt = 0; mt < 4; ++mt)
            #pragma unroll
            for (int j = 0; j < 8; ++j) {
                int c = mt * 16 + g * 8 + j;
                out[base0 + (long)c * 262144 + e] = acc[mt][j] * scale;
            }
    }
}

extern "C" void kernel_launch(void* const* d_in, const int* in_sizes, int n_in,
                              void* d_out, int out_size, void* d_ws, size_t ws_size,
                              hipStream_t stream) {
    (void)in_sizes; (void)n_in; (void)out_size; (void)ws_size;
    const float* x      = (const float*)d_in[0];
    const float* conv_w = (const float*)d_in[1];
    const float* conv_b = (const float*)d_in[2];
    const float* lin_w  = (const float*)d_in[3];
    const float* lin_b  = (const float*)d_in[4];

    __half* linw_h = (__half*)d_ws;  // 64 KB f16 copy of lin_w

    convert_linw_kernel<<<128, 256, 0, stream>>>(lin_w, linw_h);
    onn_fused_kernel<<<16 * 512, 256, 0, stream>>>(x, conv_w, conv_b, linw_h, lin_b,
                                                   (float*)d_out);
}
